// Embed2GraphEnsemble_53420803228025
// MI455X (gfx1250) — compile-verified, matrix operand order checked by reading, not run on hardware
//
#include <hip/hip_runtime.h>
#include <math.h>

// Problem constants from the reference: B=16, N=256, D=32.
#define BB 16
#define NN 256
#define DD 32

typedef __attribute__((ext_vector_type(16))) _Float16 v16h;
typedef __attribute__((ext_vector_type(8)))  float    v8f;

union HFrag { v16h v; _Float16 h[16]; };

// Kernel 1: one block per (batch, 16-row i-tile).
//  - stage weights in LDS
//  - S[j][:] = x[j] @ W_out[0:32]  (sender transform, all 256 j)
//  - R[i][:] = x[i0+i] @ W_out[32:64] (receiver transform, 16 rows)
//  - Gram tile G[16 x 256] = X_itile @ X^T via v_wmma_f32_16x16x32_f16
//  - per pair: e = relu(relu(S_j+R_i+b_out)·W_cat + b_cat); fuse with Gram via
//    the 2->8->1 sigmoid MLP; add per-row (over j) mean; write m; accumulate
//    per-batch sum / sumsq with float atomics for the layernorm pass.
__global__ __launch_bounds__(256) void e2g_main(
    const float* __restrict__ x,     const float* __restrict__ W_out,
    const float* __restrict__ b_out, const float* __restrict__ W_cat,
    const float* __restrict__ b_cat, const float* __restrict__ W_f1,
    const float* __restrict__ b_f1,  const float* __restrict__ W_f2,
    const float* __restrict__ b_f2,  float* __restrict__ out,
    float* __restrict__ stats)
{
    __shared__ float sS[NN * 33];      // stride 33 -> conflict-free (33 odd)
    __shared__ float sG[16 * NN];      // Gram tile rows i0..i0+15
    __shared__ float sR[16 * DD];
    __shared__ float sWout[64 * DD];
    __shared__ float sBout[DD], sWcat[DD], sWf1[16], sBf1[8], sWf2[8];
    __shared__ float sRed[16];

    const int tid  = threadIdx.x;
    const int lane = tid & 31;
    const int wave = tid >> 5;             // 8 waves
    const int b    = blockIdx.x >> 4;
    const int i0   = (blockIdx.x & 15) << 4;
    const float* xb = x + b * (NN * DD);

    // ---- stage weights ----
    for (int idx = tid; idx < 64 * DD; idx += 256) sWout[idx] = W_out[idx];
    if (tid < 32) { sBout[tid] = b_out[tid]; sWcat[tid] = W_cat[tid]; }
    if (tid < 16) sWf1[tid] = W_f1[tid];
    if (tid < 8)  { sBf1[tid] = b_f1[tid]; sWf2[tid] = W_f2[tid]; }
    __syncthreads();

    // ---- S: thread t owns j = t ----
    {
        float xr[32];
        #pragma unroll
        for (int k = 0; k < 32; ++k) xr[k] = xb[tid * 32 + k];
        #pragma unroll 4
        for (int d = 0; d < 32; ++d) {
            float acc = 0.f;
            #pragma unroll
            for (int k = 0; k < 32; ++k) acc += xr[k] * sWout[k * 32 + d];
            sS[tid * 33 + d] = acc;
        }
    }
    // ---- R: 16 rows x 32 cols = 512 tasks ----
    for (int task = tid; task < 512; task += 256) {
        int i = task >> 5, d = task & 31;
        float acc = 0.f;
        #pragma unroll
        for (int k = 0; k < 32; ++k)
            acc += xb[(i0 + i) * 32 + k] * sWout[(32 + k) * 32 + d];
        sR[i * 32 + d] = acc;
    }

    // ---- Gram tile via WMMA: wave w does j-tiles {w, w+8} ----
    {
        const int hlf = lane >> 4, r = lane & 15;
        // A fragment: rows i0+r, K map per ISA 16-bit A 16x32 layout
        HFrag a;
        const int abase = (i0 + r) * 32 + hlf * 8;
        #pragma unroll
        for (int w = 0; w < 8; ++w) {
            a.h[w]     = (_Float16)xb[abase + w];
            a.h[w + 8] = (_Float16)xb[abase + 16 + w];
        }
        for (int t = 0; t < 2; ++t) {
            const int j0 = (wave + t * 8) << 4;
            HFrag bm;                      // B = X^T, loaded column-major
            const int bbase = (j0 + r) * 32 + hlf * 8;
            #pragma unroll
            for (int w = 0; w < 8; ++w) {
                bm.h[w]     = (_Float16)xb[bbase + w];
                bm.h[w + 8] = (_Float16)xb[bbase + 16 + w];
            }
            v8f c = {};
            c = __builtin_amdgcn_wmma_f32_16x16x32_f16(
                    false, a.v, false, bm.v, (short)0, c, false, false);
            #pragma unroll
            for (int e = 0; e < 8; ++e)
                sG[(e + 8 * hlf) * NN + j0 + r] = c[e];  // M=e+8*hlf, N=r
        }
    }
    __syncthreads();

    const float bcat = b_cat[0];
    const float bf2  = b_f2[0];
    float lsum = 0.f, lsq = 0.f;

    for (int i = 0; i < 16; ++i) {
        const int irow = i0 + i;
        const int j = tid;
        // e = relu( relu(S_j + R_i + b_out) . W_cat + b_cat )
        float acc = bcat;
        #pragma unroll 8
        for (int d = 0; d < 32; ++d) {
            float hv = sS[j * 33 + d] + sR[i * 32 + d] + sBout[d];
            acc += fmaxf(hv, 0.f) * sWcat[d];
        }
        float e     = fmaxf(acc, 0.f);
        float mlin  = (irow == j) ? 0.f : e;      // off-diagonal mask
        float mprod = sG[i * NN + j];
        // fusion MLP 2 -> 8 -> 1, sigmoid gate
        float v = bf2;
        #pragma unroll
        for (int o = 0; o < 8; ++o) {
            float u = mlin * sWf1[o] + mprod * sWf1[8 + o] + sBf1[o];
            v += fmaxf(u, 0.f) * sWf2[o];
        }
        float wgt = 1.f / (1.f + __expf(-v));
        float m = wgt * mlin + (1.f - wgt) * mprod;

        // row mean over j (axis=-2): full row lives in this block
        float rs = m;
        #pragma unroll
        for (int off = 16; off > 0; off >>= 1) rs += __shfl_xor(rs, off, 32);
        __syncthreads();
        if (lane == 0) sRed[wave] = rs;
        __syncthreads();
        float rowsum = 0.f;
        #pragma unroll
        for (int wv = 0; wv < 8; ++wv) rowsum += sRed[wv];

        float mm = m + rowsum * (1.f / 256.f);
        out[(b * NN + irow) * NN + j] = mm;
        lsum += mm; lsq += mm * mm;
    }

    // per-batch sum / sumsq for layernorm
    #pragma unroll
    for (int off = 16; off > 0; off >>= 1) {
        lsum += __shfl_xor(lsum, off, 32);
        lsq  += __shfl_xor(lsq,  off, 32);
    }
    __syncthreads();
    if (lane == 0) { sRed[wave] = lsum; sRed[8 + wave] = lsq; }
    __syncthreads();
    if (tid == 0) {
        float ts = 0.f, tq = 0.f;
        #pragma unroll
        for (int wv = 0; wv < 8; ++wv) { ts += sRed[wv]; tq += sRed[8 + wv]; }
        atomicAdd(&stats[b], ts);
        atomicAdd(&stats[16 + b], tq);
    }
}

// Kernel 2: per-batch layernorm over all N*N elements, in place on d_out.
__global__ __launch_bounds__(256) void e2g_norm(float* __restrict__ out,
                                                const float* __restrict__ stats)
{
    const int idx = blockIdx.x * 256 + threadIdx.x;   // 1,048,576 total
    const int b   = idx >> 16;                        // 65536 elems per batch
    const float mu  = stats[b] * (1.f / 65536.f);
    const float var = stats[16 + b] * (1.f / 65536.f) - mu * mu;
    out[idx] = (out[idx] - mu) * rsqrtf(var + 1e-5f);
}

extern "C" void kernel_launch(void* const* d_in, const int* in_sizes, int n_in,
                              void* d_out, int out_size, void* d_ws, size_t ws_size,
                              hipStream_t stream) {
    (void)in_sizes; (void)n_in; (void)out_size; (void)ws_size;
    const float* x     = (const float*)d_in[0];
    const float* W_out = (const float*)d_in[1];
    const float* b_out = (const float*)d_in[2];
    const float* W_cat = (const float*)d_in[3];
    const float* b_cat = (const float*)d_in[4];
    const float* W_f1  = (const float*)d_in[5];
    const float* b_f1  = (const float*)d_in[6];
    const float* W_f2  = (const float*)d_in[7];
    const float* b_f2  = (const float*)d_in[8];
    float* out   = (float*)d_out;
    float* stats = (float*)d_ws;   // [0:16) sums, [16:32) sumsq

    hipMemsetAsync(stats, 0, 32 * sizeof(float), stream);
    e2g_main<<<BB * (NN / 16), 256, 0, stream>>>(x, W_out, b_out, W_cat, b_cat,
                                                 W_f1, b_f1, W_f2, b_f2, out, stats);
    e2g_norm<<<(BB * NN * NN) / 256, 256, 0, stream>>>(out, stats);
}
